// AttnBlock_76089640616305
// MI455X (gfx1250) — matrix operand order repaired and checked by reading, not compiled, not run on hardware
//
#include <hip/hip_runtime.h>

typedef __attribute__((ext_vector_type(16))) __bf16 v16bf;
typedef __attribute__((ext_vector_type(8)))  __bf16 v8bf;
typedef __attribute__((ext_vector_type(8)))  float  v8f;
typedef unsigned int u32x4 __attribute__((ext_vector_type(4)));
typedef int          i32x8 __attribute__((ext_vector_type(8)));
typedef int          i32x4 __attribute__((ext_vector_type(4)));

#define NB 2
#define CCH 512
#define NPIX 4096
#define NGRP 32

__device__ __forceinline__ v16bf cat16(v8bf lo, v8bf hi) {
  return __builtin_shufflevector(lo, hi, 0, 1, 2, 3, 4, 5, 6, 7,
                                 8, 9, 10, 11, 12, 13, 14, 15);
}

// ---------------------------------------------------------------------------
// TDM: issue a 2-D tensor_load_to_lds.
//   dim0 = contiguous K run (elements, 2B each), dim1 = rows (stride in elems)
// D# bitfields per CDNA5 ISA. Tiles are always
// fully in-range so tensor_dim == tile_dim. Tracked by TENSORcnt.
// Toolchain uses the 6-arg builtin (g0 v4u, g1 v8i, g2 v4i, g3 v4i, v8i, cpol).
// ---------------------------------------------------------------------------
__device__ __forceinline__ void tdm_load_2d(unsigned lds_byte_addr,
                                            const void* gaddr,
                                            unsigned tile_k,      // elems in dim0
                                            unsigned tile_rows,   // dim1
                                            unsigned long long row_stride_elems)
{
  unsigned long long ga = (unsigned long long)(uintptr_t)gaddr;
  u32x4 g0;
  g0[0] = 1u;                                         // count=1 (user D#)
  g0[1] = lds_byte_addr;                              // lds_addr
  g0[2] = (unsigned)(ga & 0xffffffffu);               // global_addr[31:0]
  g0[3] = (unsigned)((ga >> 32) & 0x1ffffffu) | (2u << 30);  // addr[56:32] | type=2

  unsigned td0 = tile_k, td1 = tile_rows;
  i32x8 g1;
  g1[0] = (int)(1u << 16);                            // wg_mask=0, data_size=1 (2B)
  g1[1] = (int)((td0 & 0xffffu) << 16);               // abar=0 | tensor_dim0[15:0]
  g1[2] = (int)(((td0 >> 16) & 0xffffu) | ((td1 & 0xffffu) << 16));
  g1[3] = (int)(((td1 >> 16) & 0xffffu) | ((tile_k & 0xffffu) << 16));   // tile_dim0
  g1[4] = (int)(tile_rows & 0xffffu);                 // tile_dim1 | tile_dim2=0
  g1[5] = (int)(unsigned)(row_stride_elems & 0xffffffffu);   // dim0_stride lo
  g1[6] = (int)(unsigned)((row_stride_elems >> 32) & 0xffffu); // hi | dim1_stride=0
  g1[7] = 0;

  i32x4 z4 = {0, 0, 0, 0};
  i32x8 z8 = {0, 0, 0, 0, 0, 0, 0, 0};
  __builtin_amdgcn_tensor_load_to_lds(g0, g1, z4, z4, z8, 0);
}

// ---------------------------------------------------------------------------
// GroupNorm: one (batch, group) = 16 ch x 4096 px. Writes h as [B][N][C] bf16
// so downstream B-operands have contiguous K.
// ---------------------------------------------------------------------------
__global__ void groupnorm_bf16_kernel(const float* __restrict__ x,
                                      const float* __restrict__ gamma,
                                      const float* __restrict__ beta,
                                      __bf16* __restrict__ h_nc)
{
  const int b = blockIdx.y;
  const int grp = blockIdx.x;
  const int chpg = CCH / NGRP;                       // 16
  const long long base = ((long long)b * CCH + (long long)grp * chpg) * NPIX;
  const int cnt = chpg * NPIX;                       // 65536

  float s = 0.f, ss = 0.f;
  for (int i = threadIdx.x; i < cnt; i += blockDim.x) {
    float v = x[base + i];
    s += v; ss += v * v;
  }
  __shared__ float r0[256], r1[256];
  r0[threadIdx.x] = s; r1[threadIdx.x] = ss;
  __syncthreads();
  for (int st = 128; st > 0; st >>= 1) {
    if ((int)threadIdx.x < st) {
      r0[threadIdx.x] += r0[threadIdx.x + st];
      r1[threadIdx.x] += r1[threadIdx.x + st];
    }
    __syncthreads();
  }
  const float mean = r0[0] / (float)cnt;
  const float var  = r1[0] / (float)cnt - mean * mean;
  const float inv  = rsqrtf(var + 1e-6f);

  for (int i = threadIdx.x; i < cnt; i += blockDim.x) {
    int cl = i / NPIX, n = i % NPIX;
    int c = grp * chpg + cl;
    float v = (x[base + i] - mean) * inv * gamma[c] + beta[c];
    h_nc[((long long)b * NPIX + n) * CCH + c] = (__bf16)v;
  }
}

// ---------------------------------------------------------------------------
__global__ void f32_to_bf16_kernel(const float* __restrict__ src,
                                   __bf16* __restrict__ dst, int n)
{
  int i = blockIdx.x * blockDim.x + threadIdx.x;
  if (i < n) dst[i] = (__bf16)src[i];
}

// ---------------------------------------------------------------------------
// bf16 WMMA GEMM with TDM->LDS double-buffered staging.
//   A[m][k]  row-major (k contiguous, stride a_rs)
//   Bt[n][k] row-major (k contiguous, stride bt_rs)   == B transposed
// Block = 8 waves (2m x 4n) -> 64x256 output tile.
// Per 32-K step: wave0 issues 2 TDM loads (A panel 64x32, B panel 256x32)
// into LDS[buf^1] while all waves compute from LDS[buf] via ds_load_b128.
// Fragment maps (this kernel's contiguous k order, same for A and B):
//   A frag: lane(l16=m,g) -> k = g*16+[0..15]
//   B frag: lane(l16=n,g) -> k = g*16+[0..15]
// ---------------------------------------------------------------------------
__global__ void gemm_bf16_wmma_kernel(
    const __bf16* __restrict__ A, long long a_rs, long long a_bstride,
    const __bf16* __restrict__ Bt, long long bt_rs, long long bt_bstride,
    int K, float alpha,
    const float* __restrict__ bias,
    const float* __restrict__ residual, long long r_bstride,
    float* __restrict__ outF, long long of_bstride,
    __bf16* __restrict__ outB, long long ob_bstride,
    long long o_rs, long long o_cs)
{
  __shared__ __bf16 sA[2][64 * 32];    //  4 KB per buffer
  __shared__ __bf16 sB[2][256 * 32];   // 16 KB per buffer

  const int lane = threadIdx.x & 31;
  const int wave = threadIdx.x >> 5;
  const int wm = wave >> 2, wn = wave & 3;
  const int bz = blockIdx.z;
  const int g = lane >> 4;
  const int l16 = lane & 15;

  const int mB = blockIdx.y * 64;      // block row base
  const int nB = blockIdx.x * 256;     // block col base

  const __bf16* Ab  = A  + (long long)bz * a_bstride + (long long)mB * a_rs;
  const __bf16* Btb = Bt + (long long)bz * bt_bstride + (long long)nB * bt_rs;

  // LDS fragment offsets (elements); row pitch 32 halves = 64 B
  const int aoff0 = (wm * 32 + l16) * 32 + g * 16;
  const int aoff1 = aoff0 + 16 * 32;
  const int boff  = ((wn * 4) * 16 + l16) * 32 + g * 16;

  // prologue: stage K-step 0 into buffer 0
  if (wave == 0) {
    tdm_load_2d((unsigned)(uintptr_t)&sA[0][0], Ab, 32, 64, (unsigned long long)a_rs);
    tdm_load_2d((unsigned)(uintptr_t)&sB[0][0], Btb, 32, 256, (unsigned long long)bt_rs);
  }

  v8f acc[2][4] = {};
  int cur = 0;
  for (int k0 = 0; k0 < K; k0 += 32) {
    if (wave == 0) __builtin_amdgcn_s_wait_tensorcnt(0);
    __syncthreads();                         // LDS[cur] ready for everyone
    if (wave == 0 && k0 + 32 < K) {          // stage next step into other buffer
      tdm_load_2d((unsigned)(uintptr_t)&sA[cur ^ 1][0], Ab + (k0 + 32), 32, 64,
                  (unsigned long long)a_rs);
      tdm_load_2d((unsigned)(uintptr_t)&sB[cur ^ 1][0], Btb + (k0 + 32), 32, 256,
                  (unsigned long long)bt_rs);
    }

    v16bf A0 = cat16(*(const v8bf*)&sA[cur][aoff0], *(const v8bf*)&sA[cur][aoff0 + 8]);
    v16bf A1 = cat16(*(const v8bf*)&sA[cur][aoff1], *(const v8bf*)&sA[cur][aoff1 + 8]);
    v16bf B0 = cat16(*(const v8bf*)&sB[cur][boff],            *(const v8bf*)&sB[cur][boff + 8]);
    v16bf B1 = cat16(*(const v8bf*)&sB[cur][boff + 16 * 32],  *(const v8bf*)&sB[cur][boff + 16 * 32 + 8]);
    v16bf B2 = cat16(*(const v8bf*)&sB[cur][boff + 32 * 32],  *(const v8bf*)&sB[cur][boff + 32 * 32 + 8]);
    v16bf B3 = cat16(*(const v8bf*)&sB[cur][boff + 48 * 32],  *(const v8bf*)&sB[cur][boff + 48 * 32 + 8]);

    acc[0][0] = __builtin_amdgcn_wmma_f32_16x16x32_bf16(false, A0, false, B0, (short)0, acc[0][0], false, false);
    acc[1][0] = __builtin_amdgcn_wmma_f32_16x16x32_bf16(false, A1, false, B0, (short)0, acc[1][0], false, false);
    acc[0][1] = __builtin_amdgcn_wmma_f32_16x16x32_bf16(false, A0, false, B1, (short)0, acc[0][1], false, false);
    acc[1][1] = __builtin_amdgcn_wmma_f32_16x16x32_bf16(false, A1, false, B1, (short)0, acc[1][1], false, false);
    acc[0][2] = __builtin_amdgcn_wmma_f32_16x16x32_bf16(false, A0, false, B2, (short)0, acc[0][2], false, false);
    acc[1][2] = __builtin_amdgcn_wmma_f32_16x16x32_bf16(false, A1, false, B2, (short)0, acc[1][2], false, false);
    acc[0][3] = __builtin_amdgcn_wmma_f32_16x16x32_bf16(false, A0, false, B3, (short)0, acc[0][3], false, false);
    acc[1][3] = __builtin_amdgcn_wmma_f32_16x16x32_bf16(false, A1, false, B3, (short)0, acc[1][3], false, false);

    __syncthreads();                         // all reads of LDS[cur] done
    cur ^= 1;
  }

  // D layout: vgpr r -> m = g*8 + r, col = l16
#pragma unroll
  for (int a = 0; a < 2; ++a) {
#pragma unroll
    for (int t = 0; t < 4; ++t) {
#pragma unroll
      for (int r = 0; r < 8; ++r) {
        int m = mB + wm * 32 + a * 16 + g * 8 + r;
        int n = nB + (wn * 4 + t) * 16 + l16;
        float val = acc[a][t][r] * alpha;
        if (bias) val += bias[m];
        long long off = (long long)m * o_rs + (long long)n * o_cs;
        if (residual) val += residual[(long long)bz * r_bstride + off];
        if (outF) outF[(long long)bz * of_bstride + off] = val;
        if (outB) outB[(long long)bz * ob_bstride + off] = (__bf16)val;
      }
    }
  }
}

// ---------------------------------------------------------------------------
__global__ void softmax_bf16_kernel(const float* __restrict__ s,
                                    __bf16* __restrict__ a)
{
  const long long row = blockIdx.x;
  const float* sr = s + row * NPIX;
  __bf16* ar = a + row * NPIX;
  __shared__ float red[256];

  float mx = -3.4e38f;
  for (int j = threadIdx.x; j < NPIX; j += 256) mx = fmaxf(mx, sr[j]);
  red[threadIdx.x] = mx; __syncthreads();
  for (int st = 128; st > 0; st >>= 1) {
    if ((int)threadIdx.x < st)
      red[threadIdx.x] = fmaxf(red[threadIdx.x], red[threadIdx.x + st]);
    __syncthreads();
  }
  mx = red[0]; __syncthreads();

  float sum = 0.f;
  for (int j = threadIdx.x; j < NPIX; j += 256) sum += __expf(sr[j] - mx);
  red[threadIdx.x] = sum; __syncthreads();
  for (int st = 128; st > 0; st >>= 1) {
    if ((int)threadIdx.x < st) red[threadIdx.x] += red[threadIdx.x + st];
    __syncthreads();
  }
  const float inv = 1.f / red[0];
  for (int j = threadIdx.x; j < NPIX; j += 256)
    ar[j] = (__bf16)(__expf(sr[j] - mx) * inv);
}

// ---------------------------------------------------------------------------
extern "C" void kernel_launch(void* const* d_in, const int* in_sizes, int n_in,
                              void* d_out, int out_size, void* d_ws, size_t ws_size,
                              hipStream_t stream) {
  (void)in_sizes; (void)n_in; (void)out_size; (void)ws_size;
  const float* x     = (const float*)d_in[0];
  const float* gamma = (const float*)d_in[1];
  const float* beta  = (const float*)d_in[2];
  const float* wq = (const float*)d_in[3]; const float* bq = (const float*)d_in[4];
  const float* wk = (const float*)d_in[5]; const float* bk = (const float*)d_in[6];
  const float* wv = (const float*)d_in[7]; const float* bv = (const float*)d_in[8];
  const float* wo = (const float*)d_in[9]; const float* bo = (const float*)d_in[10];
  float* out = (float*)d_out;

  const long long CN = (long long)CCH * NPIX;

  size_t off = 0;
  auto take = [&](size_t bytes) {
    void* p = (char*)d_ws + off;
    off += (bytes + 255) & ~(size_t)255;
    return p;
  };
  __bf16* h_nc  = (__bf16*)take((size_t)NB * CN * 2);      // [B][N][C]
  __bf16* wq_bf = (__bf16*)take((size_t)CCH * CCH * 2);
  __bf16* wk_bf = (__bf16*)take((size_t)CCH * CCH * 2);
  __bf16* wv_bf = (__bf16*)take((size_t)CCH * CCH * 2);
  __bf16* wo_bf = (__bf16*)take((size_t)CCH * CCH * 2);
  __bf16* q_nc  = (__bf16*)take((size_t)NB * CN * 2);      // [B][N][C]
  __bf16* k_nc  = (__bf16*)take((size_t)NB * CN * 2);      // [B][N][C]
  __bf16* v_cn  = (__bf16*)take((size_t)NB * CN * 2);      // [B][C][N]
  __bf16* o_nc  = (__bf16*)take((size_t)NB * CN * 2);      // [B][N][C]
  float*  scores = (float*)take((size_t)NPIX * NPIX * 4);  // per-batch [N][N]
  __bf16* a_bf   = (__bf16*)take((size_t)NPIX * NPIX * 2); // per-batch [N][N]

  // 1) GroupNorm -> h_nc
  groupnorm_bf16_kernel<<<dim3(NGRP, NB), 256, 0, stream>>>(x, gamma, beta, h_nc);

  // 2) weights -> bf16
  {
    int n = CCH * CCH, blocks = (n + 255) / 256;
    f32_to_bf16_kernel<<<blocks, 256, 0, stream>>>(wq, wq_bf, n);
    f32_to_bf16_kernel<<<blocks, 256, 0, stream>>>(wk, wk_bf, n);
    f32_to_bf16_kernel<<<blocks, 256, 0, stream>>>(wv, wv_bf, n);
    f32_to_bf16_kernel<<<blocks, 256, 0, stream>>>(wo, wo_bf, n);
  }

  const dim3 blk(256);
  const dim3 grid_cn(NPIX / 256, CCH / 64, NB);   // M=512,  N=4096, batched
  const dim3 grid_nn(NPIX / 256, NPIX / 64, 1);   // M=4096, N=4096, one batch
  const dim3 grid_cn1(NPIX / 256, CCH / 64, 1);

  // 3) q/k = W*h -> [N][C] ; v = W*h -> [C][N]
  gemm_bf16_wmma_kernel<<<grid_cn, blk, 0, stream>>>(
      wq_bf, CCH, 0,  h_nc, CCH, CN,  CCH, 1.0f, bq,
      nullptr, 0,  nullptr, 0,  q_nc, CN,  1, CCH);
  gemm_bf16_wmma_kernel<<<grid_cn, blk, 0, stream>>>(
      wk_bf, CCH, 0,  h_nc, CCH, CN,  CCH, 1.0f, bk,
      nullptr, 0,  nullptr, 0,  k_nc, CN,  1, CCH);
  gemm_bf16_wmma_kernel<<<grid_cn, blk, 0, stream>>>(
      wv_bf, CCH, 0,  h_nc, CCH, CN,  CCH, 1.0f, bv,
      nullptr, 0,  nullptr, 0,  v_cn, CN,  NPIX, 1);

  const float scale = 0.04419417382415922f;  // 512^-0.5

  // 4) per-batch attention
  for (int b = 0; b < NB; ++b) {
    const __bf16* qb = q_nc + (long long)b * CN;
    const __bf16* kb = k_nc + (long long)b * CN;
    const __bf16* vb = v_cn + (long long)b * CN;
    __bf16* ob = o_nc + (long long)b * CN;

    // scores[i,j] = scale * sum_c q_nc[i,c] * k_nc[j,c]
    gemm_bf16_wmma_kernel<<<grid_nn, blk, 0, stream>>>(
        qb, CCH, 0,  kb, CCH, 0,  CCH, scale, nullptr,
        nullptr, 0,  scores, 0,  nullptr, 0,  NPIX, 1);

    softmax_bf16_kernel<<<dim3(NPIX), 256, 0, stream>>>(scores, a_bf);

    // o[c,i] = sum_j v_cn[c,j] * a[i,j]
    gemm_bf16_wmma_kernel<<<grid_cn1, blk, 0, stream>>>(
        vb, NPIX, 0,  a_bf, NPIX, 0,  NPIX, 1.0f, nullptr,
        nullptr, 0,  nullptr, 0,  ob, 0,  1, CCH);
  }

  // 5) out = x + Wo*o + bo
  gemm_bf16_wmma_kernel<<<grid_cn, blk, 0, stream>>>(
      wo_bf, CCH, 0,  o_nc, CCH, CN,  CCH, 1.0f, bo,
      x, CN,  out, CN,  nullptr, 0,  NPIX, 1);
}
